// SelfAttention_489626271788
// MI455X (gfx1250) — matrix operand order repaired and checked by reading, not compiled, not run on hardware
//
#include <hip/hip_runtime.h>

// Flash-attention forward for B=2,H=12,S=2048,D=64 (fp32 in/out) on gfx1250.
// Matmuls in f16 WMMA with f32 accumulation; softmax fully in f32.

typedef _Float16 v16h __attribute__((ext_vector_type(16)));
typedef _Float16 v8h  __attribute__((ext_vector_type(8)));
typedef float    v8f  __attribute__((ext_vector_type(8)));

#define SEQ 2048
#define DIM 64
#define NH  12
#define BM  128        // query rows per block (8 waves x 16 rows)
#define BN  64         // key rows per iteration
#define LDK 72         // padded LDS row stride in halves (144B rows: 16B aligned, 36-dword bank stride)
#define NEG_INF (-1000000000.0f)

union AFrag { v16h v; v8h h8[2]; };

__global__ __launch_bounds__(256)
void fa_fwd_f16wmma(const float* __restrict__ Q, const float* __restrict__ K,
                    const float* __restrict__ V, const int* __restrict__ M,
                    float* __restrict__ O)
{
    __shared__ __attribute__((aligned(16))) _Float16 sQ[BM * LDK];
    __shared__ __attribute__((aligned(16))) _Float16 sK[BN * LDK];      // [j][d]
    __shared__ __attribute__((aligned(16))) _Float16 sV[DIM * LDK];     // transposed: [d][j]
    __shared__ __attribute__((aligned(16))) _Float16 sP[8][16 * LDK];   // per-wave P tile

    const int tid  = threadIdx.x;
    const int wid  = tid >> 5;
    const int lane = tid & 31;
    const int nl   = lane & 15;     // N (or M) index within 16
    const int hh   = lane >> 4;     // lane half

    const int qb = blockIdx.x;      // query block index
    const int bh = blockIdx.y;      // fused batch*head
    const int b  = bh / NH;

    const float* Qg = Q + ((size_t)bh * SEQ + (size_t)qb * BM) * DIM;
    const float* Kg = K + (size_t)bh * SEQ * DIM;
    const float* Vg = V + (size_t)bh * SEQ * DIM;
    const int*   Mg = M + (size_t)b * SEQ * SEQ + ((size_t)qb * BM + (size_t)wid * 16) * SEQ;
    float*       Og = O + ((size_t)bh * SEQ + (size_t)qb * BM + (size_t)wid * 16) * DIM;

    // ---- stage Q tile (128x64) into LDS as f16, coalesced f32 reads ----
    #pragma unroll
    for (int e = 0; e < (BM * DIM) / 256; ++e) {
        int idx = e * 256 + tid;
        int r = idx >> 6, d = idx & 63;
        sQ[r * LDK + d] = (_Float16)Qg[(size_t)r * DIM + d];
    }
    __syncthreads();

    // ---- build Q A-fragments: 16x32 f16, contiguous k order (matches B-frags) ----
    // lane m = l%16; element s: K = 16*(l/16)+s
    AFrag aq[2];
    {
        const _Float16* qrow = &sQ[(wid * 16 + nl) * LDK];
        #pragma unroll
        for (int df = 0; df < 2; ++df) {
            aq[df].h8[0] = *(const v8h*)(qrow + df * 32 + 16 * hh);
            aq[df].h8[1] = *(const v8h*)(qrow + df * 32 + 16 * hh + 8);
        }
    }

    v8f acc[4] = {};                 // O accumulator: 4 d-tiles of 16x16 f32 (C layout)
    float mrow[8], lrow[8];          // per-row running max / sum; row = 8*hh + r
    #pragma unroll
    for (int r = 0; r < 8; ++r) { mrow[r] = -INFINITY; lrow[r] = 0.0f; }

    for (int kb = 0; kb < SEQ; kb += BN) {
        __syncthreads();             // previous iteration done reading sK/sV
        // ---- cooperative K/V tile load, f32 -> f16, V transposed ----
        #pragma unroll
        for (int e = 0; e < (BN * DIM) / 256; ++e) {
            int idx = e * 256 + tid;
            int j = idx >> 6, d = idx & 63;
            float kv = Kg[(size_t)(kb + j) * DIM + d];
            float vv = Vg[(size_t)(kb + j) * DIM + d];
            sK[j * LDK + d] = (_Float16)kv;
            sV[d * LDK + j] = (_Float16)vv;
        }
        __syncthreads();

        // ---- GEMM1: S(16x64) = Q(16x64) . K^T ; B-frag = contiguous 32B of K row ----
        v8f st[4];
        #pragma unroll
        for (int t = 0; t < 4; ++t) {
            v8f c = {};
            #pragma unroll
            for (int kh = 0; kh < 2; ++kh) {
                v16h bf = *(const v16h*)&sK[(t * 16 + nl) * LDK + kh * 32 + 16 * hh];
                c = __builtin_amdgcn_wmma_f32_16x16x32_f16(false, aq[kh].v, false, bf,
                                                           (short)0, c, false, false);
            }
            st[t] = c;
        }

        // ---- scale + mask (C layout: VGPR r -> row 8*hh+r, lane nl -> col) ----
        #pragma unroll
        for (int t = 0; t < 4; ++t) {
            #pragma unroll
            for (int r = 0; r < 8; ++r) {
                float s = st[t][r] * 0.125f;  // 1/sqrt(64)
                int mk = Mg[(size_t)(8 * hh + r) * SEQ + kb + t * 16 + nl];
                st[t][r] = (mk != 0) ? s : NEG_INF;
            }
        }

        // ---- online softmax: reductions across the 16 lanes sharing each row ----
        #pragma unroll
        for (int r = 0; r < 8; ++r) {
            float mx = fmaxf(fmaxf(st[0][r], st[1][r]), fmaxf(st[2][r], st[3][r]));
            mx = fmaxf(mx, __shfl_xor(mx, 1, 32));
            mx = fmaxf(mx, __shfl_xor(mx, 2, 32));
            mx = fmaxf(mx, __shfl_xor(mx, 4, 32));
            mx = fmaxf(mx, __shfl_xor(mx, 8, 32));
            float mnew  = fmaxf(mrow[r], mx);
            float alpha = __expf(mrow[r] - mnew);
            mrow[r] = mnew;
            float rs = 0.0f;
            #pragma unroll
            for (int t = 0; t < 4; ++t) {
                float p = __expf(st[t][r] - mnew);
                st[t][r] = p;
                rs += p;
            }
            rs += __shfl_xor(rs, 1, 32);
            rs += __shfl_xor(rs, 2, 32);
            rs += __shfl_xor(rs, 4, 32);
            rs += __shfl_xor(rs, 8, 32);
            lrow[r] = lrow[r] * alpha + rs;
            #pragma unroll
            for (int dt = 0; dt < 4; ++dt) acc[dt][r] *= alpha;
        }

        // ---- P -> f16 in per-wave LDS region (C layout -> memory), then reload as A-frags ----
        _Float16* myP = &sP[wid][0];
        #pragma unroll
        for (int t = 0; t < 4; ++t) {
            #pragma unroll
            for (int r = 0; r < 8; ++r)
                myP[(8 * hh + r) * LDK + t * 16 + nl] = (_Float16)st[t][r];
        }
        asm volatile("s_wait_dscnt 0x0" ::: "memory");  // in-wave LDS RAW fence

        AFrag ap[2];
        #pragma unroll
        for (int jf = 0; jf < 2; ++jf) {
            const _Float16* prow = myP + nl * LDK + jf * 32;
            ap[jf].h8[0] = *(const v8h*)(prow + 16 * hh);
            ap[jf].h8[1] = *(const v8h*)(prow + 16 * hh + 8);
        }

        // ---- GEMM2: O += P(16x64) . V(64x64) ; B-frag = contiguous 32B of V^T row ----
        #pragma unroll
        for (int dt = 0; dt < 4; ++dt) {
            #pragma unroll
            for (int jf = 0; jf < 2; ++jf) {
                v16h bf = *(const v16h*)&sV[(dt * 16 + nl) * LDK + jf * 32 + 16 * hh];
                acc[dt] = __builtin_amdgcn_wmma_f32_16x16x32_f16(false, ap[jf].v, false, bf,
                                                                 (short)0, acc[dt], false, false);
            }
        }
    }

    // ---- epilogue: normalize by row sum, store fp32 coalesced ----
    #pragma unroll
    for (int dt = 0; dt < 4; ++dt) {
        #pragma unroll
        for (int r = 0; r < 8; ++r) {
            Og[(size_t)(8 * hh + r) * DIM + dt * 16 + nl] = acc[dt][r] / lrow[r];
        }
    }
}

extern "C" void kernel_launch(void* const* d_in, const int* in_sizes, int n_in,
                              void* d_out, int out_size, void* d_ws, size_t ws_size,
                              hipStream_t stream) {
    (void)n_in; (void)out_size; (void)d_ws; (void)ws_size;
    const float* q = (const float*)d_in[0];
    const float* k = (const float*)d_in[1];
    const float* v = (const float*)d_in[2];
    const int*   m = (const int*)d_in[3];
    float* out = (float*)d_out;

    const int BH = in_sizes[0] / (SEQ * DIM);   // B*H = 24
    dim3 grid(SEQ / BM, BH);
    dim3 block(256);
    fa_fwd_f16wmma<<<grid, block, 0, stream>>>(q, k, v, m, out);
}
